// TransformerEncoder_16071767621745
// MI455X (gfx1250) — matrix operand order repaired and checked by reading, not compiled, not run on hardware
//
#include <hip/hip_runtime.h>
#include <math.h>

// ---------------------------------------------------------------------------
// Types for CDNA5 WMMA (wave32)
// ---------------------------------------------------------------------------
typedef __bf16 bf16_t;
typedef __bf16 v16bf __attribute__((ext_vector_type(16)));
typedef __bf16 v8bf  __attribute__((ext_vector_type(8)));
typedef float  v8f   __attribute__((ext_vector_type(8)));

struct v8bf_pair { v8bf lo, hi; };

static __device__ __forceinline__ v16bf make_v16(v8bf lo, v8bf hi) {
    v8bf_pair p{lo, hi};
    return __builtin_bit_cast(v16bf, p);
}

// A-fragment (16x32 bf16, MxK): lane holds row m; elements 0..15 = K[koff..koff+15],
// koff = hi*16, contiguous.  Two 16B LDS loads.
static __device__ __forceinline__ v16bf lds_afrag(const bf16_t* base, int row,
                                                  int stride, int kb, int hi) {
    const bf16_t* p = base + row * stride + kb + hi * 16;
    v8bf lo = *(const v8bf*)p;
    v8bf hh = *(const v8bf*)(p + 8);
    return make_v16(lo, hh);
}

// B-fragment (32x16 bf16, KxN): lane holds col n; elements 0..15 = K[koff..koff+15],
// koff = hi*16, contiguous.  Two 16B LDS loads.
static __device__ __forceinline__ v16bf lds_bfrag(const bf16_t* base, int col,
                                                  int stride, int kb, int hi) {
    const bf16_t* p = base + col * stride + kb + hi * 16;
    v8bf lo = *(const v8bf*)p;
    v8bf hh = *(const v8bf*)(p + 8);
    return make_v16(lo, hh);
}

static __device__ __forceinline__ v8f wmma_bf16(v16bf a, v16bf b, v8f c) {
    return __builtin_amdgcn_wmma_f32_16x16x32_bf16(
        /*neg_a=*/false, a, /*neg_b=*/false, b,
        /*c_mod=*/(short)0, c, /*reuse_a=*/false, /*reuse_b=*/false);
}

// CDNA5 async global->LDS 16B copy (bypasses VGPRs, tracked on ASYNCcnt).
// SADDR form: uniform 64-bit base (SGPR pair) + per-lane 32-bit byte offset;
// LDS destination byte address = low 32 bits of the generic LDS pointer.
static __device__ __forceinline__ void async_copy16(const void* gbase,
                                                    unsigned byte_off,
                                                    void* lds_dst) {
    unsigned lds_addr = (unsigned)(size_t)lds_dst;
    asm volatile("global_load_async_to_lds_b128 %0, %1, %2"
                 :
                 : "v"(lds_addr), "v"(byte_off),
                   "s"((unsigned long long)(size_t)gbase)
                 : "memory");
}

// Wait until at most N async copies remain in flight (per-wave ASYNCcnt).
static __device__ __forceinline__ void wait_async_le4() {
    asm volatile("s_wait_asynccnt 0x4" ::: "memory");
}
static __device__ __forceinline__ void wait_async_le0() {
    asm volatile("s_wait_asynccnt 0x0" ::: "memory");
}

static __device__ __forceinline__ float gelu_exact(float x) {
    return 0.5f * x * (1.0f + erff(x * 0.70710678118654752f));
}

// Problem constants
#define NB   16
#define NT   1024
#define ND   768
#define NH   12
#define NHD  64
#define NMLP 3072

// ---------------------------------------------------------------------------
// Weight prep: W[K,N] f32 -> Wt[N,K] bf16 (LDS-tiled 32x32 transpose)
// ---------------------------------------------------------------------------
__global__ __launch_bounds__(256) void transpose_w_kernel(
    const float* __restrict__ in, bf16_t* __restrict__ out, int K, int N) {
    __shared__ float tile[32][33];
    const int nb = blockIdx.x * 32, kb = blockIdx.y * 32;
    const int tx = threadIdx.x & 31, ty = threadIdx.x >> 5;   // 32 x 8
#pragma unroll
    for (int i = 0; i < 4; ++i) {
        int r = ty + i * 8;
        tile[r][tx] = in[(size_t)(kb + r) * N + nb + tx];
    }
    __syncthreads();
#pragma unroll
    for (int i = 0; i < 4; ++i) {
        int r = ty + i * 8;                 // local n
        out[(size_t)(nb + r) * K + kb + tx] = (bf16_t)tile[tx][r];
    }
}

// ---------------------------------------------------------------------------
// LayerNorm: one 256-thread block per row of 768 f32 -> bf16 out
// ---------------------------------------------------------------------------
__global__ __launch_bounds__(256) void layernorm_bf16_kernel(
    const float* __restrict__ x, const float* __restrict__ g,
    const float* __restrict__ b, bf16_t* __restrict__ out) {
    const int row  = blockIdx.x;
    const int tid  = threadIdx.x;
    const int lane = tid & 31;
    const int w    = tid >> 5;
    const float* xr = x + (size_t)row * ND;

    float s = 0.f, s2 = 0.f;
    float vals[3];
#pragma unroll
    for (int i = 0; i < 3; ++i) {
        float v = xr[tid + i * 256];
        vals[i] = v;
        s += v; s2 += v * v;
    }
#pragma unroll
    for (int off = 16; off >= 1; off >>= 1) {
        s  += __shfl_xor(s,  off, 32);
        s2 += __shfl_xor(s2, off, 32);
    }
    __shared__ float ws1[8], ws2[8];
    if (lane == 0) { ws1[w] = s; ws2[w] = s2; }
    __syncthreads();
    if (tid == 0) {
        float a = 0.f, a2 = 0.f;
#pragma unroll
        for (int i = 0; i < 8; ++i) { a += ws1[i]; a2 += ws2[i]; }
        ws1[0] = a; ws2[0] = a2;
    }
    __syncthreads();
    const float mean = ws1[0] * (1.0f / ND);
    const float var  = ws2[0] * (1.0f / ND) - mean * mean;
    const float rstd = rsqrtf(var + 1e-5f);
    bf16_t* orow = out + (size_t)row * ND;
#pragma unroll
    for (int i = 0; i < 3; ++i) {
        int c = tid + i * 256;
        orow[c] = (bf16_t)((vals[i] - mean) * rstd * g[c] + b[c]);
    }
}

// ---------------------------------------------------------------------------
// bf16 WMMA GEMM:  dst = epilogue((A[M,K] @ Wt[N,K]^T + bias) * oscale)
// A and Wt both bf16.  Block tile 128x128, K-step 32, 8 waves (4M x 2N),
// each wave 32x64 (2x4 WMMA tiles).  Both tiles staged with async global->LDS
// copies, double-buffered: tile kb+1 issued before waiting on tile kb
// (s_wait_asynccnt 4 leaves the next tile's 4 copies in flight).
// MODE 0: bf16 store to [B,H,T,HD]           (q with oscale=1/sqrt(D), k)
// MODE 1: bf16 store to [B,H,HD,T]           (v, transposed for P@V B-frags)
// MODE 2: f32  store  acc + bias + resid     (attn proj + residual)
// MODE 3: bf16 store  gelu(acc + bias)       (MLP fc1)
// MODE 4: f32  store  gelu(acc+bias) + resid (MLP fc2 + residual -> d_out)
// ---------------------------------------------------------------------------
template <int MODE>
__global__ __launch_bounds__(256, 2) void gemm_bf16_kernel(
    const bf16_t* __restrict__ A, const bf16_t* __restrict__ Wt,
    const float* __restrict__ bias, const float* __restrict__ resid,
    void* __restrict__ dst, int M, int N, int K, float oscale) {
    __shared__ bf16_t As[2][128 * 32];   // [m][k], k contiguous
    __shared__ bf16_t Bs[2][128 * 32];   // [n][k], k contiguous

    const int tid  = threadIdx.x;
    const int lane = tid & 31;
    const int w    = tid >> 5;
    const int wm   = w >> 1;          // 0..3
    const int wn   = w & 1;           // 0..1
    const int hi   = lane >> 4;
    const int fr   = lane & 15;
    const int rowbase = blockIdx.x * 128;
    const int colbase = blockIdx.y * 128;

    const bf16_t* Ag = A  + (size_t)rowbase * K;
    const bf16_t* Wg = Wt + (size_t)colbase * K;
    // Per-thread chunk coordinates (128 rows x 32 k = 512 16B-chunks, 2/thread)
    int crow[2], ckc[2];
#pragma unroll
    for (int i = 0; i < 2; ++i) {
        int c   = i * 256 + tid;
        crow[i] = c >> 2;
        ckc[i]  = (c & 3) * 8;
    }

    auto stage = [&](int buf, int kb) {
#pragma unroll
        for (int i = 0; i < 2; ++i) {
            unsigned off = (unsigned)(((size_t)crow[i] * K + kb * 32 + ckc[i]) *
                                      sizeof(bf16_t));
            async_copy16(Ag, off, As[buf] + crow[i] * 32 + ckc[i]);
            async_copy16(Wg, off, Bs[buf] + crow[i] * 32 + ckc[i]);
        }
    };

    v8f acc[2][4] = {};
    const int ksteps = K >> 5;

    stage(0, 0);
    for (int kb = 0; kb < ksteps; ++kb) {
        const int cur = kb & 1;
        if (kb + 1 < ksteps) {
            stage(cur ^ 1, kb + 1);
            wait_async_le4();            // tile kb landed; kb+1 in flight
        } else {
            wait_async_le0();
        }
        __syncthreads();                 // tile kb visible to all waves

        v16bf af[2], bf[4];
#pragma unroll
        for (int mt = 0; mt < 2; ++mt)
            af[mt] = lds_afrag(As[cur], wm * 32 + mt * 16 + fr, 32, 0, hi);
#pragma unroll
        for (int nt = 0; nt < 4; ++nt)
            bf[nt] = lds_bfrag(Bs[cur], wn * 64 + nt * 16 + fr, 32, 0, hi);
#pragma unroll
        for (int mt = 0; mt < 2; ++mt)
#pragma unroll
            for (int nt = 0; nt < 4; ++nt)
                acc[mt][nt] = wmma_bf16(af[mt], bf[nt], acc[mt][nt]);

        __syncthreads();                 // done reading buf before it's refilled
    }

    // Epilogue.  C layout: VGPR r -> row (r + 8*hi), col = fr
#pragma unroll
    for (int mt = 0; mt < 2; ++mt) {
#pragma unroll
        for (int nt = 0; nt < 4; ++nt) {
#pragma unroll
            for (int r = 0; r < 8; ++r) {
                int gm = rowbase + wm * 32 + mt * 16 + r + 8 * hi;
                int gn = colbase + wn * 64 + nt * 16 + fr;
                float v = acc[mt][nt][r] + bias[gn];
                if (MODE == 0 || MODE == 1) {
                    v *= oscale;
                    int b = gm >> 10, t = gm & 1023;        // T = 1024
                    int hh = gn >> 6, hd = gn & 63;         // HD = 64
                    bf16_t* o = (bf16_t*)dst;
                    if (MODE == 0)
                        o[(((size_t)(b * NH + hh) * NT + t) << 6) + hd] = (bf16_t)v;
                    else
                        o[(((size_t)(b * NH + hh) * NHD + hd) << 10) + t] = (bf16_t)v;
                } else if (MODE == 2) {
                    ((float*)dst)[(size_t)gm * N + gn] =
                        v + resid[(size_t)gm * N + gn];
                } else if (MODE == 3) {
                    ((bf16_t*)dst)[(size_t)gm * N + gn] = (bf16_t)gelu_exact(v);
                } else { // MODE 4
                    ((float*)dst)[(size_t)gm * N + gn] =
                        gelu_exact(v) + resid[(size_t)gm * N + gn];
                }
            }
        }
    }
}

// ---------------------------------------------------------------------------
// Attention (flash-style, online softmax).  Grid: (T/128, B*H).  8 waves/block,
// each wave owns 16 query rows.  q (pre-scaled by 1/sqrt(D)), k: [B,H,T,HD];
// v: [B,H,HD,T] bf16.  K/V chunks double-buffered on ASYNCcnt.
// Output: attn [B,T,D] bf16.
// ---------------------------------------------------------------------------
__global__ __launch_bounds__(256, 1) void attention_kernel(
    const bf16_t* __restrict__ q, const bf16_t* __restrict__ k,
    const bf16_t* __restrict__ v, bf16_t* __restrict__ attn) {
    __shared__ bf16_t Qs[128 * 64];       // [q_local][hd]
    __shared__ bf16_t Ks[2][64 * 64];     // [key_local][hd]
    __shared__ bf16_t Vs[2][64 * 64];     // [hd][key_local]  (V^T tile)
    __shared__ bf16_t Ps[8][16 * 64];     // per-wave P tile [q_row][key_local]

    const int tid  = threadIdx.x;
    const int lane = tid & 31;
    const int w    = tid >> 5;
    const int hi   = lane >> 4;
    const int fr   = lane & 15;
    const int bh    = blockIdx.y;         // b*H + h
    const int qbase = blockIdx.x * 128;

    const bf16_t* qg = q + (size_t)bh * NT * NHD + (size_t)qbase * NHD;
    const bf16_t* kg = k + (size_t)bh * NT * NHD;
    const bf16_t* vg = v + (size_t)bh * NHD * NT;

    // Stage Q block tile (128x64 bf16 = 16KB) via async global->LDS
#pragma unroll
    for (int i = 0; i < 4; ++i) {
        int c   = i * 256 + tid;          // 0..1023 chunks of 16B
        int row = c >> 3;                 // 8 chunks per row
        int kc  = (c & 7) * 8;
        async_copy16(qg, (unsigned)(((size_t)row * NHD + kc) * sizeof(bf16_t)),
                     Qs + row * 64 + kc);
    }
    wait_async_le0();
    __syncthreads();

    v16bf aq[2];
#pragma unroll
    for (int kk = 0; kk < 2; ++kk)
        aq[kk] = lds_afrag(Qs, w * 16 + fr, 64, kk * 32, hi);

    // Per-thread chunk coordinates for K/V staging (512 chunks, 2/thread each)
    int crow[2], ckc[2];
#pragma unroll
    for (int i = 0; i < 2; ++i) {
        int c   = i * 256 + tid;
        crow[i] = c >> 3;
        ckc[i]  = (c & 7) * 8;
    }
    auto stage_kv = [&](int buf, int ch) {
#pragma unroll
        for (int i = 0; i < 2; ++i) {
            async_copy16(kg,
                (unsigned)(((size_t)(ch * 64 + crow[i]) * NHD + ckc[i]) * sizeof(bf16_t)),
                Ks[buf] + crow[i] * 64 + ckc[i]);
            async_copy16(vg,
                (unsigned)(((size_t)crow[i] * NT + ch * 64 + ckc[i]) * sizeof(bf16_t)),
                Vs[buf] + crow[i] * 64 + ckc[i]);
        }
    };

    float mloc[8], lsum[8];
    v8f o[4] = {};
#pragma unroll
    for (int r = 0; r < 8; ++r) { mloc[r] = -1e30f; lsum[r] = 0.f; }

    const int nch = NT / 64;
    stage_kv(0, 0);
    for (int ch = 0; ch < nch; ++ch) {
        const int cur = ch & 1;
        if (ch + 1 < nch) {
            stage_kv(cur ^ 1, ch + 1);
            wait_async_le4();             // chunk ch landed; ch+1 in flight
        } else {
            wait_async_le0();
        }
        __syncthreads();

        // S = Q @ K^T for 16 queries x 64 keys (4 tiles); q already scaled
        v8f s[4];
#pragma unroll
        for (int t = 0; t < 4; ++t) {
            v16bf b0 = lds_bfrag(Ks[cur], t * 16 + fr, 64, 0, hi);
            v16bf b1 = lds_bfrag(Ks[cur], t * 16 + fr, 64, 32, hi);
            v8f a = {};
            a = wmma_bf16(aq[0], b0, a);
            a = wmma_bf16(aq[1], b1, a);
            s[t] = a;
        }

        // per-row chunk max
        float cmax[8];
#pragma unroll
        for (int r = 0; r < 8; ++r)
            cmax[r] = fmaxf(fmaxf(s[0][r], s[1][r]), fmaxf(s[2][r], s[3][r]));
#pragma unroll
        for (int off = 8; off >= 1; off >>= 1)
#pragma unroll
            for (int r = 0; r < 8; ++r)
                cmax[r] = fmaxf(cmax[r], __shfl_xor(cmax[r], off, 32));

        float alpha[8], rs[8];
#pragma unroll
        for (int r = 0; r < 8; ++r) {
            float mn = fmaxf(mloc[r], cmax[r]);
            alpha[r] = __expf(mloc[r] - mn);
            mloc[r]  = mn;
            rs[r]    = 0.f;
        }
        // P = exp(S - m), write bf16 tile to per-wave LDS in A layout (row-major)
#pragma unroll
        for (int t = 0; t < 4; ++t)
#pragma unroll
            for (int r = 0; r < 8; ++r) {
                float pv = __expf(s[t][r] - mloc[r]);
                rs[r] += pv;
                Ps[w][(r + 8 * hi) * 64 + t * 16 + fr] = (bf16_t)pv;
            }
#pragma unroll
        for (int off = 8; off >= 1; off >>= 1)
#pragma unroll
            for (int r = 0; r < 8; ++r)
                rs[r] += __shfl_xor(rs[r], off, 32);
#pragma unroll
        for (int r = 0; r < 8; ++r)
            lsum[r] = lsum[r] * alpha[r] + rs[r];
#pragma unroll
        for (int t = 0; t < 4; ++t)
#pragma unroll
            for (int r = 0; r < 8; ++r)
                o[t][r] *= alpha[r];

        // same-wave LDS RAW: ensure P stores complete before fragment reloads
        asm volatile("s_wait_dscnt 0x0" ::: "memory");

        v16bf ap[2];
#pragma unroll
        for (int kk = 0; kk < 2; ++kk)
            ap[kk] = lds_afrag(Ps[w], fr, 64, kk * 32, hi);
#pragma unroll
        for (int t = 0; t < 4; ++t) {   // t = hd tile
            v16bf b0 = lds_bfrag(Vs[cur], t * 16 + fr, 64, 0, hi);
            v16bf b1 = lds_bfrag(Vs[cur], t * 16 + fr, 64, 32, hi);
            o[t] = wmma_bf16(ap[0], b0, o[t]);
            o[t] = wmma_bf16(ap[1], b1, o[t]);
        }
        __syncthreads();                 // done reading cur buf before refill
    }

    // Normalize and write attn [B,T,D] bf16 (D index = h*HD + hd)
    const int b = bh / NH, h = bh % NH;
#pragma unroll
    for (int t = 0; t < 4; ++t)
#pragma unroll
        for (int r = 0; r < 8; ++r) {
            int trow = qbase + w * 16 + r + 8 * hi;
            int d    = h * NHD + t * 16 + fr;
            attn[((size_t)(b * NT + trow)) * ND + d] =
                (bf16_t)(o[t][r] / lsum[r]);
        }
}

// ---------------------------------------------------------------------------
// Host launcher
// ---------------------------------------------------------------------------
extern "C" void kernel_launch(void* const* d_in, const int* in_sizes, int n_in,
                              void* d_out, int out_size, void* d_ws, size_t ws_size,
                              hipStream_t stream) {
    const float* x     = (const float*)d_in[0];
    const float* Wq    = (const float*)d_in[1];
    const float* bq    = (const float*)d_in[2];
    const float* Wk    = (const float*)d_in[3];
    const float* bk    = (const float*)d_in[4];
    const float* Wv    = (const float*)d_in[5];
    const float* bv    = (const float*)d_in[6];
    const float* Wo    = (const float*)d_in[7];
    const float* bo    = (const float*)d_in[8];
    const float* ln1_g = (const float*)d_in[9];
    const float* ln1_b = (const float*)d_in[10];
    const float* ln2_g = (const float*)d_in[11];
    const float* ln2_b = (const float*)d_in[12];
    const float* W1    = (const float*)d_in[13];
    const float* b1    = (const float*)d_in[14];
    const float* W2    = (const float*)d_in[15];
    const float* b2    = (const float*)d_in[16];

    const size_t Mtok = (size_t)NB * NT;           // 16384
    char* p = (char*)d_ws;
    auto alloc = [&](size_t bytes) {
        char* r = p;
        p += (bytes + 255) & ~(size_t)255;
        return r;
    };
    bf16_t* hbf   = (bf16_t*)alloc(Mtok * ND * 2);
    bf16_t* qb    = (bf16_t*)alloc(Mtok * ND * 2);
    bf16_t* kb_   = (bf16_t*)alloc(Mtok * ND * 2);
    bf16_t* vb    = (bf16_t*)alloc(Mtok * ND * 2);
    bf16_t* attnb = (bf16_t*)alloc(Mtok * ND * 2);
    float*  out1  = (float*)alloc(Mtok * ND * 4);
    bf16_t* h2    = (bf16_t*)alloc(Mtok * ND * 2);
    bf16_t* mb    = (bf16_t*)alloc(Mtok * NMLP * 2);
    bf16_t* WqT   = (bf16_t*)alloc((size_t)ND * ND * 2);
    bf16_t* WkT   = (bf16_t*)alloc((size_t)ND * ND * 2);
    bf16_t* WvT   = (bf16_t*)alloc((size_t)ND * ND * 2);
    bf16_t* WoT   = (bf16_t*)alloc((size_t)ND * ND * 2);
    bf16_t* W1T   = (bf16_t*)alloc((size_t)ND * NMLP * 2);
    bf16_t* W2T   = (bf16_t*)alloc((size_t)ND * NMLP * 2);

    const dim3 blk(256);
    const dim3 gD((int)(Mtok / 128), ND / 128);     // (128, 6)
    const dim3 gM((int)(Mtok / 128), NMLP / 128);   // (128, 24)
    const dim3 gTd(ND / 32, ND / 32);               // 768x768 transpose
    const dim3 gT1(NMLP / 32, ND / 32);             // K=768, N=3072
    const dim3 gT2(ND / 32, NMLP / 32);             // K=3072, N=768
    const float qscale = 0.03608439182435161f;      // 1/sqrt(768)

    // 0) Convert + transpose all weights to bf16 [N,K] (reused 128x each)
    transpose_w_kernel<<<gTd, blk, 0, stream>>>(Wq, WqT, ND, ND);
    transpose_w_kernel<<<gTd, blk, 0, stream>>>(Wk, WkT, ND, ND);
    transpose_w_kernel<<<gTd, blk, 0, stream>>>(Wv, WvT, ND, ND);
    transpose_w_kernel<<<gTd, blk, 0, stream>>>(Wo, WoT, ND, ND);
    transpose_w_kernel<<<gT1, blk, 0, stream>>>(W1, W1T, ND, NMLP);
    transpose_w_kernel<<<gT2, blk, 0, stream>>>(W2, W2T, NMLP, ND);

    // 1) LN1 -> h (bf16)
    layernorm_bf16_kernel<<<(int)Mtok, blk, 0, stream>>>(x, ln1_g, ln1_b, hbf);
    // 2-4) Q (pre-scaled), K, V projections (V stored transposed per-head)
    gemm_bf16_kernel<0><<<gD, blk, 0, stream>>>(hbf, WqT, bq, nullptr, qb,
                                                (int)Mtok, ND, ND, qscale);
    gemm_bf16_kernel<0><<<gD, blk, 0, stream>>>(hbf, WkT, bk, nullptr, kb_,
                                                (int)Mtok, ND, ND, 1.0f);
    gemm_bf16_kernel<1><<<gD, blk, 0, stream>>>(hbf, WvT, bv, nullptr, vb,
                                                (int)Mtok, ND, ND, 1.0f);
    // 5) attention
    attention_kernel<<<dim3(NT / 128, NB * NH), blk, 0, stream>>>(qb, kb_, vb, attnb);
    // 6) output projection + residual -> out1 (f32)
    gemm_bf16_kernel<2><<<gD, blk, 0, stream>>>(attnb, WoT, bo, x, out1,
                                                (int)Mtok, ND, ND, 1.0f);
    // 7) LN2 -> h2 (bf16)
    layernorm_bf16_kernel<<<(int)Mtok, blk, 0, stream>>>(out1, ln2_g, ln2_b, h2);
    // 8) MLP fc1 + GELU -> m (bf16)
    gemm_bf16_kernel<3><<<gM, blk, 0, stream>>>(h2, W1T, b1, nullptr, mb,
                                                (int)Mtok, NMLP, ND, 1.0f);
    // 9) MLP fc2 + GELU + residual -> d_out (f32)
    gemm_bf16_kernel<4><<<gD, blk, 0, stream>>>(mb, W2T, b2, out1, (float*)d_out,
                                                (int)Mtok, ND, NMLP, 1.0f);
}
